// Support_model_CA_78529182040412
// MI455X (gfx1250) — compile-verified, matrix operand order checked by reading, not run on hardware
//
#include <hip/hip_runtime.h>

// ---------------------------------------------------------------------------
// Problem constants (fixed by the reference)
// ---------------------------------------------------------------------------
#define BB   8
#define MMG  128          // h nodes per graph
#define RRG  256          // ret nodes per graph
#define NBR  16
#define NH   (BB*MMG)     // 1024
#define NRET (BB*RRG)     // 2048
#define EE   (NH*NBR)     // 16384 edges
#define E3   (3*EE)       // 49152 (planar vector rows)
#define NG   64
#define CUT_R 10.0f

#define FLAG_ACC 1
#define FLAG_SIG 2

typedef __attribute__((ext_vector_type(16))) __bf16 v16bf;
typedef __attribute__((ext_vector_type(8)))  float  v8f;

// ---------------------------------------------------------------------------
// Generic WMMA GEMM:  C[M,N] = act( scale*(A[M,K] @ B[N,K]^T + bias) (+Cin) )
// A rows optionally gathered through rowidx. M%64==0, N%64==0, K%32==0.
// One wave computes a 64x64 macro tile as 4x4 WMMA 16x16x32 bf16 tiles,
// converting f32 operands to bf16 in-register (f32 accumulate).
// Fragment layouts follow CDNA5 ISA 7.12.2 (16-bit A 16x32; C/D 16x16).
// ---------------------------------------------------------------------------
__global__ __launch_bounds__(32)
void wmma_gemm_kernel(const float* __restrict__ A, int lda,
                      const int* __restrict__ rowidx,
                      const float* __restrict__ B, int ldb,
                      const float* __restrict__ bias,
                      const float* Cin,
                      float* C, int ldc,
                      int K, float scale, int flags)
{
    const int lane = threadIdx.x;
    const int lo = lane & 15;
    const int hi = lane >> 4;
    const int m0 = blockIdx.y * 64;
    const int n0 = blockIdx.x * 64;

    const float* Arow[4];
    const float* Brow[4];
#pragma unroll
    for (int t = 0; t < 4; ++t) {
        int r = m0 + t * 16 + lo;
        if (rowidx) r = rowidx[r];
        Arow[t] = A + (size_t)r * lda;
        Brow[t] = B + (size_t)(n0 + t * 16 + lo) * ldb;
    }

    v8f acc[16];
#pragma unroll
    for (int i = 0; i < 16; ++i) acc[i] = (v8f){};

    for (int k0 = 0; k0 < K; k0 += 32) {
        v16bf af[4], bf[4];
#pragma unroll
        for (int t = 0; t < 4; ++t) {
#pragma unroll
            for (int j = 0; j < 8; ++j) {
                // k within 32-step: lanes0-15: vgpr0-3 -> k0..7, vgpr4-7 -> k16..23
                //                   lanes16-31: +8
                int kk = k0 + ((j & 4) << 2) + hi * 8 + (j & 3) * 2;
                float2 av = *(const float2*)(Arow[t] + kk);
                float2 bv = *(const float2*)(Brow[t] + kk);
                af[t][2 * j]     = (__bf16)av.x;
                af[t][2 * j + 1] = (__bf16)av.y;
                bf[t][2 * j]     = (__bf16)bv.x;
                bf[t][2 * j + 1] = (__bf16)bv.y;
            }
        }
#pragma unroll
        for (int tm = 0; tm < 4; ++tm)
#pragma unroll
            for (int tn = 0; tn < 4; ++tn)
                acc[tm * 4 + tn] = __builtin_amdgcn_wmma_f32_16x16x32_bf16(
                    false, af[tm], false, bf[tn], (short)0, acc[tm * 4 + tn],
                    false, false);
    }

#pragma unroll
    for (int tm = 0; tm < 4; ++tm) {
#pragma unroll
        for (int tn = 0; tn < 4; ++tn) {
            int col = n0 + tn * 16 + lo;
            float bb = bias ? bias[col] : 0.0f;
#pragma unroll
            for (int r = 0; r < 8; ++r) {
                int row = m0 + tm * 16 + r + hi * 8;   // C/D layout: lanes16-31 -> M+8
                float v = (acc[tm * 4 + tn][r] + bb) * scale;
                if (flags & FLAG_ACC) v += Cin[(size_t)row * ldc + col];
                if (flags & FLAG_SIG) v = 1.0f / (1.0f + __expf(-v));
                C[(size_t)row * ldc + col] = v;
            }
        }
    }
}

// ---------------------------------------------------------------------------
// kNN: one block per h-node. 256 threads compute d2 to the 256 ret nodes of
// the node's graph, then 16 argmin rounds (selection order is irrelevant:
// edge outputs are summed per node downstream).
// ---------------------------------------------------------------------------
__global__ __launch_bounds__(256)
void knn_kernel(const float* __restrict__ h_pos, const float* __restrict__ ret_pos,
                int* __restrict__ gidx)
{
    __shared__ float d2[RRG];
    __shared__ float rv[RRG];
    __shared__ int   ri[RRG];
    const int n = blockIdx.x;
    const int t = threadIdx.x;
    const int b = n >> 7;                    // graph id (MMG = 128)
    const float hx = h_pos[n * 3 + 0], hy = h_pos[n * 3 + 1], hz = h_pos[n * 3 + 2];
    const float* rp = ret_pos + (size_t)(b * RRG + t) * 3;
    const float dx = hx - rp[0], dy = hy - rp[1], dz = hz - rp[2];
    d2[t] = dx * dx + dy * dy + dz * dz;
    __syncthreads();
    for (int it = 0; it < NBR; ++it) {
        rv[t] = d2[t]; ri[t] = t;
        __syncthreads();
        for (int s = 128; s > 0; s >>= 1) {
            if (t < s && rv[t + s] < rv[t]) { rv[t] = rv[t + s]; ri[t] = ri[t + s]; }
            __syncthreads();
        }
        if (t == 0) {
            int best = ri[0];
            gidx[n * NBR + it] = b * RRG + best;
            d2[best] = 3.0e38f;
        }
        __syncthreads();
    }
}

// ---------------------------------------------------------------------------
// Edge features + vector gather: one thread per (edge, channel).
// Writes Gaussian smearing, planar edge direction expansion, planar gathered
// ret_vec, and the cosine cutoff.
// ---------------------------------------------------------------------------
__global__ void edge_feat_kernel(const float* __restrict__ h_pos,
                                 const float* __restrict__ ret_pos,
                                 const float* __restrict__ ret_vec,
                                 const int* __restrict__ gidx,
                                 const float* __restrict__ eeW,
                                 float* __restrict__ edge_sca,
                                 float* __restrict__ edge_vec_p,
                                 float* __restrict__ new_vec_p,
                                 float* __restrict__ Ccut)
{
    int idx = blockIdx.x * blockDim.x + threadIdx.x;
    if (idx >= EE * NG) return;
    const int e = idx >> 6, c = idx & 63;
    const int n = e >> 4;                    // NBR = 16
    const int g = gidx[e];
    const float dx = h_pos[n * 3 + 0] - ret_pos[g * 3 + 0];
    const float dy = h_pos[n * 3 + 1] - ret_pos[g * 3 + 1];
    const float dz = h_pos[n * 3 + 2] - ret_pos[g * 3 + 2];
    const float dist = sqrtf(dx * dx + dy * dy + dz * dz);

    const float step = CUT_R / (float)(NG - 1);
    const float diff = dist - (float)c * step;
    edge_sca[idx] = __expf((-0.5f / (step * step)) * diff * diff);

    const float inv = 1.0f / (dist + 1e-7f);
    const float w = eeW[c];
    const size_t plane = (size_t)EE * NG;
    edge_vec_p[0 * plane + idx] = dx * inv * w;
    edge_vec_p[1 * plane + idx] = dy * inv * w;
    edge_vec_p[2 * plane + idx] = dz * inv * w;

    const float* rv = ret_vec + ((size_t)g * NG + c) * 3;
    new_vec_p[0 * plane + idx] = rv[0];
    new_vec_p[1 * plane + idx] = rv[1];
    new_vec_p[2 * plane + idx] = rv[2];

    if (c == 0) {
        float Cc = 0.5f * (__cosf(dist * 3.14159265358979f / CUT_R) + 1.0f);
        Ccut[e] = (dist <= CUT_R) ? Cc : 0.0f;
    }
}

// ---- small elementwise kernels -------------------------------------------
__global__ void norm3_kernel(const float* __restrict__ v, float* __restrict__ out)
{
    int idx = blockIdx.x * blockDim.x + threadIdx.x;
    if (idx >= EE * NG) return;
    const size_t plane = (size_t)EE * NG;
    float a = v[idx], b = v[plane + idx], c = v[2 * plane + idx];
    out[idx] = sqrtf(a * a + b * b + c * c);
}

__global__ void gatemul_kernel(float* __restrict__ v, const float* __restrict__ gate)
{
    int idx = blockIdx.x * blockDim.x + threadIdx.x;
    if (idx >= EE * NG) return;
    const size_t plane = (size_t)EE * NG;
    float g = gate[idx];
    v[idx] *= g; v[plane + idx] *= g; v[2 * plane + idx] *= g;
}

__global__ void vnleaky_kernel(float* __restrict__ x, const float* __restrict__ d)
{
    int idx = blockIdx.x * blockDim.x + threadIdx.x;
    if (idx >= EE * NG) return;
    const size_t plane = (size_t)EE * NG;
    float x0 = x[idx], x1 = x[plane + idx], x2 = x[2 * plane + idx];
    float d0 = d[idx], d1 = d[plane + idx], d2 = d[2 * plane + idx];
    float dot = x0 * d0 + x1 * d1 + x2 * d2;
    float dsq = d0 * d0 + d1 * d1 + d2 * d2;
    float t = (dot >= 0.0f) ? 0.0f : (dot / (dsq + 1e-6f));
    x[idx]             = 0.2f * x0 + 0.8f * (x0 - t * d0);
    x[plane + idx]     = 0.2f * x1 + 0.8f * (x1 - t * d1);
    x[2 * plane + idx] = 0.2f * x2 + 0.8f * (x2 - t * d2);
}

__global__ void leaky_kernel(float* __restrict__ x, int n)
{
    int idx = blockIdx.x * blockDim.x + threadIdx.x;
    if (idx >= n) return;
    float v = x[idx];
    x[idx] = (v >= 0.0f) ? v : 0.01f * v;
}

__global__ void mul_kernel(float* __restrict__ dst, const float* __restrict__ a, int n)
{
    int idx = blockIdx.x * blockDim.x + threadIdx.x;
    if (idx >= n) return;
    dst[idx] *= a[idx];
}

__global__ void yvec_kernel(float* __restrict__ y, const float* __restrict__ a1,
                            const float* __restrict__ nodevec,
                            const float* __restrict__ a2, const float* __restrict__ ev)
{
    int idx = blockIdx.x * blockDim.x + threadIdx.x;
    if (idx >= EE * NG) return;
    const size_t plane = (size_t)EE * NG;
    float g1 = a1[idx], g2 = a2[idx];
#pragma unroll
    for (int p = 0; p < 3; ++p)
        y[p * plane + idx] = g1 * nodevec[p * plane + idx] + g2 * ev[p * plane + idx];
}

// scatter (edges are contiguous per node) fused with cutoff (and gate for vec)
__global__ void scatter_sca_kernel(const float* __restrict__ osca,
                                   const float* __restrict__ Ccut,
                                   float* __restrict__ hadd)
{
    int idx = blockIdx.x * blockDim.x + threadIdx.x;
    if (idx >= NH * 256) return;
    int n = idx >> 8, c = idx & 255;
    float s = 0.0f;
#pragma unroll
    for (int j = 0; j < NBR; ++j) {
        int e = n * NBR + j;
        s += Ccut[e] * osca[(size_t)e * 256 + c];
    }
    hadd[idx] = s;
}

__global__ void scatter_vec_kernel(const float* __restrict__ ovec,
                                   const float* __restrict__ gate,
                                   const float* __restrict__ Ccut,
                                   float* __restrict__ haddr)   // layout (n*3+d, 64)
{
    int idx = blockIdx.x * blockDim.x + threadIdx.x;
    if (idx >= NH * NG) return;
    int n = idx >> 6, h = idx & 63;
    const size_t plane = (size_t)EE * NG;
#pragma unroll
    for (int d = 0; d < 3; ++d) {
        float s = 0.0f;
        for (int j = 0; j < NBR; ++j) {
            int e = n * NBR + j;
            s += Ccut[e] * gate[(size_t)e * NG + h] * ovec[d * plane + (size_t)e * NG + h];
        }
        haddr[((size_t)(n * 3 + d)) * NG + h] = s;
    }
}

__global__ void qvgather_kernel(const float* __restrict__ h_vec, float* __restrict__ qv)
{
    int idx = blockIdx.x * blockDim.x + threadIdx.x;   // over 3*NH*64
    if (idx >= NH * 3 * NG) return;
    int row = idx >> 6, v = idx & 63;
    int n = row / 3, d = row - n * 3;
    qv[idx] = h_vec[(size_t)n * (NG * 3) + v * 3 + d];
}

__global__ void outvec_kernel(const float* __restrict__ t, float* __restrict__ out)
{
    int idx = blockIdx.x * blockDim.x + threadIdx.x;   // over NH*64*3, out (n,v,d)
    if (idx >= NH * NG * 3) return;
    int n = idx / (NG * 3), r = idx - n * (NG * 3);
    int v = r / 3, d = r - v * 3;
    out[idx] = t[((size_t)(n * 3 + d)) * NG + v];
}

// ---------------------------------------------------------------------------
// Streaming-softmax attention: one wave per (query row, head). Lanes stride
// over keys with per-lane online softmax, then a wave32 butterfly merge.
// ---------------------------------------------------------------------------
template <int HD>
__global__ __launch_bounds__(32)
void attn_kernel(const float* __restrict__ Q, const float* __restrict__ Km,
                 const float* __restrict__ Vm, float* __restrict__ O,
                 int L, int emb)
{
    const int row = blockIdx.x, head = blockIdx.y, lane = threadIdx.x;
    const float* q = Q + (size_t)row * emb + head * HD;
    float qr[HD];
#pragma unroll
    for (int i = 0; i < HD; ++i) qr[i] = q[i];
    float m = -3.0e38f, l = 0.0f;
    float acc[HD];
#pragma unroll
    for (int i = 0; i < HD; ++i) acc[i] = 0.0f;

    for (int k = lane; k < L; k += 32) {
        const float* kp = Km + (size_t)k * emb + head * HD;
        const float* vp = Vm + (size_t)k * emb + head * HD;
        float s = 0.0f;
#pragma unroll
        for (int i = 0; i < HD; ++i) s += qr[i] * kp[i];
        float mn = fmaxf(m, s);
        float c0 = __expf(m - mn), w = __expf(s - mn);
        l = l * c0 + w;
#pragma unroll
        for (int i = 0; i < HD; ++i) acc[i] = acc[i] * c0 + w * vp[i];
        m = mn;
    }
#pragma unroll
    for (int off = 16; off >= 1; off >>= 1) {
        float mo = __shfl_xor(m, off, 32);
        float lo = __shfl_xor(l, off, 32);
        float mn = fmaxf(m, mo);
        float c0 = __expf(m - mn), c1 = __expf(mo - mn);
#pragma unroll
        for (int i = 0; i < HD; ++i) {
            float ao = __shfl_xor(acc[i], off, 32);
            acc[i] = acc[i] * c0 + ao * c1;
        }
        l = l * c0 + lo * c1;
        m = mn;
    }
    if (lane == 0) {
        float inv = 1.0f / l;
        float* o = O + (size_t)row * emb + head * HD;
#pragma unroll
        for (int i = 0; i < HD; ++i) o[i] = acc[i] * inv;
    }
}

// ---------------------------------------------------------------------------
// Host side
// ---------------------------------------------------------------------------
static inline void gemm(hipStream_t s, const float* A, int lda, const int* ridx,
                        const float* B, int ldb, const float* bias,
                        float* C, int ldc, int M, int N, int K,
                        float scale, int flags)
{
    dim3 grid(N / 64, M / 64);
    wmma_gemm_kernel<<<grid, 32, 0, s>>>(A, lda, ridx, B, ldb, bias, C, C, ldc,
                                         K, scale, flags);
}

static inline dim3 g1(int n) { return dim3((n + 255) / 256); }

extern "C" void kernel_launch(void* const* d_in, const int* in_sizes, int n_in,
                              void* d_out, int out_size, void* d_ws, size_t ws_size,
                              hipStream_t stream)
{
    if (n_in < 41) return;
    // ---- input mapping: setup_inputs() flattened in dict-insertion order ----
    const float* h_sca   = (const float*)d_in[0];
    const float* h_vec   = (const float*)d_in[1];
    const float* h_pos   = (const float*)d_in[2];
    // d_in[3] = h_idx (unused)
    const float* ret_sca = (const float*)d_in[4];
    const float* ret_vec = (const float*)d_in[5];
    const float* ret_pos = (const float*)d_in[6];
    // d_in[7] = ret_idx (unused)
    const float* eeW     = (const float*)d_in[8];
    const float* nWv1 = (const float*)d_in[9],  *nWv2 = (const float*)d_in[10];
    const float* nScalW = (const float*)d_in[11];
    const float* nGateW = (const float*)d_in[12], *nGateB = (const float*)d_in[13];
    const float* eWv1 = (const float*)d_in[14], *eWv2 = (const float*)d_in[15];
    const float* eScalW = (const float*)d_in[16];
    const float* eGateW = (const float*)d_in[17], *eGateB = (const float*)d_in[18];
    const float* dirW   = (const float*)d_in[19];
    const float* scaLinW = (const float*)d_in[20], *scaLinB = (const float*)d_in[21];
    const float* e2nW = (const float*)d_in[22], *e2nB = (const float*)d_in[23];
    const float* n2eW = (const float*)d_in[24], *n2eB = (const float*)d_in[25];
    const float* evnW = (const float*)d_in[26], *evnB = (const float*)d_in[27];
    const float* oWv1 = (const float*)d_in[28], *oWv2 = (const float*)d_in[29];
    const float* oScalW = (const float*)d_in[30];
    const float* oGateW = (const float*)d_in[31], *oGateB = (const float*)d_in[32];
    const float* asWin = (const float*)d_in[33], *asBin = (const float*)d_in[34];
    const float* asWo  = (const float*)d_in[35], *asBo  = (const float*)d_in[36];
    const float* avWin = (const float*)d_in[37], *avBin = (const float*)d_in[38];
    const float* avWo  = (const float*)d_in[39], *avBo  = (const float*)d_in[40];

    // ---- workspace carve (f32; all sizes even -> 8B aligned float2 loads) ----
    size_t off = 0;
    char* base = (char*)d_ws;
    auto F = [&](size_t nfloats) -> float* {
        float* p = (float*)(base + off);
        off += nfloats * sizeof(float);
        return p;
    };
    const size_t PV = (size_t)E3 * NG;   // planar vector buffer (3*E*64)
    int*   gidx   = (int*)F(EE);
    float* Ccut   = F(EE);
    float* edge_s = F((size_t)EE * NG);
    float* bufA   = F(PV);               // edge_vec_p -> dvec -> ev_p
    float* bufB   = F(PV);               // new_vec_p -> y_vec_p -> out_vec_p
    float* inter  = F(PV);               // inter_n / inter_e / inter_o
    float* nrm    = F((size_t)EE * NG);
    float* sca256 = F((size_t)EE * 256); // node out_sca, later out_gv out_sca
    float* nvec   = F(PV);               // gated node_vec
    float* gate   = F((size_t)EE * NG);  // gate_n / gate_e / gate_o
    float* esca   = F((size_t)EE * NG);
    float* evec   = F(PV);               // gated e_vec (vn_leaky in place)
    float* ysca   = F((size_t)EE * 256); // t1 -> y_sca
    float* a1     = F((size_t)EE * NG);
    float* a2     = F((size_t)EE * NG);
    float* hadds  = F((size_t)NH * 256);
    float* haddv  = F((size_t)NH * 3 * NG);
    float* qs = F((size_t)NH * 256), *ks = F((size_t)NH * 256);
    float* vs = F((size_t)NH * 256), *osb = F((size_t)NH * 256);
    float* qvs = F((size_t)NH * 3 * NG), *qp = F((size_t)NH * 3 * NG);
    float* kp  = F((size_t)NH * 3 * NG), *vp = F((size_t)NH * 3 * NG);
    float* opv = F((size_t)NH * 3 * NG), *tmpv = F((size_t)NH * 3 * NG);
    if (off > ws_size) return;

    float* out_sca = (float*)d_out;                 // (1024, 256)
    float* out_vec = (float*)d_out + NH * 256;      // (1024, 64, 3)
    const int EN = EE * NG;

    // ---- kNN + edge features -------------------------------------------------
    knn_kernel<<<NH, 256, 0, stream>>>(h_pos, ret_pos, gidx);
    edge_feat_kernel<<<g1(EN), 256, 0, stream>>>(h_pos, ret_pos, ret_vec, gidx, eeW,
                                                 edge_s, bufA, bufB, Ccut);

    // ---- node_gv (on gathered ret features) ----------------------------------
    gemm(stream, bufB, NG, nullptr, nWv1, NG, nullptr, inter, NG, E3, NG, NG, 1.f, 0);
    norm3_kernel<<<g1(EN), 256, 0, stream>>>(inter, nrm);
    gemm(stream, nrm, NG, nullptr, nScalW, 320, nullptr, sca256, 256, EE, 256, 64, 1.f, 0);
    gemm(stream, ret_sca, 256, gidx, nScalW + 64, 320, nullptr, sca256, 256, EE, 256, 256, 1.f, FLAG_ACC);
    gemm(stream, inter, NG, nullptr, nWv2, NG, nullptr, nvec, NG, E3, NG, NG, 1.f, 0);
    gemm(stream, sca256, 256, nullptr, nGateW, 256, nGateB, gate, NG, EE, NG, 256, 1.f, FLAG_SIG);
    gatemul_kernel<<<g1(EN), 256, 0, stream>>>(nvec, gate);

    // ---- edge_gv -------------------------------------------------------------
    gemm(stream, bufA, NG, nullptr, eWv1, NG, nullptr, inter, NG, E3, NG, NG, 1.f, 0);
    norm3_kernel<<<g1(EN), 256, 0, stream>>>(inter, nrm);
    gemm(stream, nrm, NG, nullptr, eScalW, 128, nullptr, esca, NG, EE, NG, 64, 1.f, 0);
    gemm(stream, edge_s, NG, nullptr, eScalW + 64, 128, nullptr, esca, NG, EE, NG, 64, 1.f, FLAG_ACC);
    gemm(stream, inter, NG, nullptr, eWv2, NG, nullptr, evec, NG, E3, NG, NG, 1.f, 0);
    gemm(stream, esca, NG, nullptr, eGateW, NG, eGateB, gate, NG, EE, NG, NG, 1.f, FLAG_SIG);
    gatemul_kernel<<<g1(EN), 256, 0, stream>>>(evec, gate);

    // ---- VN-LeakyReLU on e_vec, LeakyReLU(0.01) on e_sca ---------------------
    gemm(stream, evec, NG, nullptr, dirW, NG, nullptr, bufA, NG, E3, NG, NG, 1.f, 0);
    vnleaky_kernel<<<g1(EN), 256, 0, stream>>>(evec, bufA);
    leaky_kernel<<<g1(EN), 256, 0, stream>>>(esca, EN);

    // ---- message combine -----------------------------------------------------
    gemm(stream, esca, NG, nullptr, scaLinW, NG, scaLinB, ysca, 256, EE, 256, 64, 1.f, 0);
    mul_kernel<<<g1(EE * 256), 256, 0, stream>>>(ysca, sca256, EE * 256);     // y_sca
    gemm(stream, evec, NG, nullptr, evnW, NG, evnB, bufA, NG, E3, NG, NG, 1.f, 0); // ev
    gemm(stream, esca, NG, nullptr, e2nW, NG, e2nB, a1, NG, EE, NG, NG, 1.f, 0);
    gemm(stream, sca256, 256, nullptr, n2eW, 256, n2eB, a2, NG, EE, NG, 256, 1.f, 0);
    yvec_kernel<<<g1(EN), 256, 0, stream>>>(bufB, a1, nvec, a2, bufA);        // y_vec

    // ---- out_gv --------------------------------------------------------------
    gemm(stream, bufB, NG, nullptr, oWv1, NG, nullptr, inter, NG, E3, NG, NG, 1.f, 0);
    norm3_kernel<<<g1(EN), 256, 0, stream>>>(inter, nrm);
    gemm(stream, nrm, NG, nullptr, oScalW, 320, nullptr, sca256, 256, EE, 256, 64, 1.f, 0);
    gemm(stream, ysca, 256, nullptr, oScalW + 64, 320, nullptr, sca256, 256, EE, 256, 256, 1.f, FLAG_ACC);
    gemm(stream, inter, NG, nullptr, oWv2, NG, nullptr, bufB, NG, E3, NG, NG, 1.f, 0); // out vec (pre-gate)
    gemm(stream, sca256, 256, nullptr, oGateW, 256, oGateB, gate, NG, EE, NG, 256, 1.f, FLAG_SIG);

    // ---- cutoff + scatter add (gate applied to vec inside scatter) -----------
    scatter_sca_kernel<<<g1(NH * 256), 256, 0, stream>>>(sca256, Ccut, hadds);
    scatter_vec_kernel<<<g1(NH * NG), 256, 0, stream>>>(bufB, gate, Ccut, haddv);

    // ---- scalar cross-attention (16 heads, hd=16, global over 1024) ----------
    gemm(stream, h_sca, 256, nullptr, asWin, 256, asBin, qs, 256, NH, 256, 256, 0.25f, 0);
    gemm(stream, hadds, 256, nullptr, asWin + 256 * 256, 256, asBin + 256, ks, 256, NH, 256, 256, 1.f, 0);
    gemm(stream, hadds, 256, nullptr, asWin + 512 * 256, 256, asBin + 512, vs, 256, NH, 256, 256, 1.f, 0);
    attn_kernel<16><<<dim3(NH, 16), 32, 0, stream>>>(qs, ks, vs, osb, NH, 256);
    gemm(stream, osb, 256, nullptr, asWo, 256, asBo, out_sca, 256, NH, 256, 256, 1.f, 0);

    // ---- vector cross-attention (8 heads, hd=8, rows = 3*NH) -----------------
    qvgather_kernel<<<g1(NH * 3 * NG), 256, 0, stream>>>(h_vec, qvs);
    gemm(stream, qvs, NG, nullptr, avWin, NG, avBin, qp, NG, 3 * NH, NG, NG, 0.35355339059f, 0);
    gemm(stream, haddv, NG, nullptr, avWin + 64 * 64, NG, avBin + 64, kp, NG, 3 * NH, NG, NG, 1.f, 0);
    gemm(stream, haddv, NG, nullptr, avWin + 128 * 64, NG, avBin + 128, vp, NG, 3 * NH, NG, NG, 1.f, 0);
    attn_kernel<8><<<dim3(3 * NH, 8), 32, 0, stream>>>(qp, kp, vp, opv, 3 * NH, NG);
    gemm(stream, opv, NG, nullptr, avWo, NG, avBo, tmpv, NG, 3 * NH, NG, NG, 1.f, 0);
    outvec_kernel<<<g1(NH * NG * 3), 256, 0, stream>>>(tmpv, out_vec);
}
